// ReasoningLoop_19061064860104
// MI455X (gfx1250) — matrix operand order repaired and checked by reading, not compiled, not run on hardware
//
#include <hip/hip_runtime.h>

// ---------------------------------------------------------------------------
// Problem constants
// ---------------------------------------------------------------------------
#define D        512
#define BATCH    4096
#define STEPS    16
#define ROWS     32                  // batch rows per workgroup
#define NWG      (BATCH / ROWS)      // 128 workgroups
#define NTHREADS 256                 // 8 wave32s
#define AS       520                 // padded bf16 row stride (bank-conflict-free)

static const long long BD       = (long long)BATCH * D;      // 2,097,152
static const long long POS_OFF  = 2  * BD;                   // positions (17,B,D)
static const long long VEL_OFF  = 19 * BD;                   // velocities (17,B,D)
static const long long GZ_OFF   = 36 * BD;                   // gray_zones (17)
static const long long CONV_OFF = 36 * BD + 17;              // converged (B)

// Workspace: 3 packed bf16 weights (512 KB each) + per-WG gray-zone partials
#define WFRAG_ELEMS (512 * 512)            // ushorts per weight
#define GZPART_OFF  (3 * WFRAG_ELEMS * 2)  // byte offset of partials

typedef __attribute__((ext_vector_type(16))) __bf16 v16bf;
typedef __attribute__((ext_vector_type(8)))  __bf16 v8bf;
typedef __attribute__((ext_vector_type(8)))  float  v8f;

__device__ __forceinline__ unsigned short f2bf(float f) {
  unsigned int u = __builtin_bit_cast(unsigned int, f);
  u += 0x7FFFu + ((u >> 16) & 1u);            // round to nearest even
  return (unsigned short)(u >> 16);
}

// ---------------------------------------------------------------------------
// Prep: pack W (N=512 outputs, K=512 inputs, row-major f32, row stride
// `stride`) into bf16 B-operand fragments.
// Fragment (nt, kt) = 32x16 (KxN) tile at frag index nt*16+kt, 512 ushorts.
// Within fragment, ushort index = lane*16 + jh where in this code's contiguous
// k order (A uses the same): n = nt*16 + (lane&15), k = kt*32 + (lane>=16 ? 16 : 0) + jh.
// ---------------------------------------------------------------------------
__global__ void prep_pack(const float* __restrict__ Wv,
                          const float* __restrict__ W1,
                          const float* __restrict__ W2,
                          unsigned short* __restrict__ out) {
  const int e = blockIdx.x * NTHREADS + threadIdx.x;   // 0 .. 3*2^18-1
  const int w    = e >> 18;
  const int idx  = e & 262143;
  const int frag = idx >> 9;          // 512 fragments per weight
  const int wi   = idx & 511;
  const int lane = wi >> 4;
  const int jh   = wi & 15;
  const int nt   = frag >> 4;
  const int kt   = frag & 15;
  const int n = nt * 16 + (lane & 15);
  const int k = kt * 32 + ((lane >> 4) << 4) + jh;
  const float* W   = (w == 0) ? Wv : ((w == 1) ? W1 : W2);
  const int stride = (w == 1) ? (2 * D) : D;
  out[e] = f2bf(W[(long long)n * stride + k]);
}

// ---------------------------------------------------------------------------
// One wave: 16x128 slab = row-tile rt of the bf16 A shadow (LDS) x 8 N-tiles
// of pre-packed bf16 weights, K=512 in 16 WMMA steps of 32.
// A fragment per lane = two contiguous 8-elt runs (matches packed B):
//   lane<16 : row m=lane,    K = k0 + {0..7} and k0 + {8..15}
//   lane>=16: row m=lane-16, K = k0+16 + {0..7} and k0+16 + {8..15}
// -> two ds_load_b128 per fragment, no conversion VALU in the loop.
// ---------------------------------------------------------------------------
__device__ __forceinline__ void gemm_rowtile(
    const unsigned short* __restrict__ Abf,  // LDS bf16, ROWS x AS
    const unsigned short* __restrict__ Bf,   // packed bf16 fragments (global)
    int rt, int nt0, int lane, v8f acc[8]) {
  const int m  = rt * 16 + (lane & 15);
  const int kb = (lane >> 4) << 4;           // 0 or 16
  const unsigned short* __restrict__ row = Abf + m * AS;
  for (int kt = 0; kt < 16; ++kt) {
    const int k0 = kt * 32 + kb;
    const v8bf lo = *(const v8bf*)(row + k0);
    const v8bf hi = *(const v8bf*)(row + k0 + 8);
    const v16bf a = __builtin_shufflevector(
        lo, hi, 0, 1, 2, 3, 4, 5, 6, 7, 8, 9, 10, 11, 12, 13, 14, 15);
#pragma unroll
    for (int i = 0; i < 8; ++i) {
      const v16bf b =
          *(const v16bf*)(Bf + (((nt0 + i) * 16 + kt) << 9) + (lane << 4));
      acc[i] = __builtin_amdgcn_wmma_f32_16x16x32_bf16(
          false, a, false, b, (short)0, acc[i], false, false);
    }
  }
}

// ---------------------------------------------------------------------------
// LDS layout (float offsets):
//   xs f32[32][512] | vs f32[32][512] | xbf bf16[32][AS] | hbf bf16[32][AS] |
//   psum[256] | gate[32] | gzv[32]
// ---------------------------------------------------------------------------
#define SM_XS   0
#define SM_VS   16384
#define SM_XBF  32768                       // 32*AS ushorts = 8320 floats
#define SM_HBF  (SM_XBF + (ROWS * AS) / 2)
#define SM_PSUM (SM_HBF + (ROWS * AS) / 2)
#define SM_GATE (SM_PSUM + 256)
#define SM_GZ   (SM_GATE + 32)
#define SM_FLTS (SM_GZ + 32)

__global__ void __launch_bounds__(NTHREADS)
reasoning_main(const float* __restrict__ x0g, const float* __restrict__ noise,
               const float* __restrict__ bv,  const float* __restrict__ b1,
               const float* __restrict__ b2,
               const float* __restrict__ Wg1, const float* __restrict__ bg1,
               const float* __restrict__ Wg2, const float* __restrict__ bg2,
               const float* __restrict__ rdecay, const float* __restrict__ rdt,
               const float* __restrict__ rspring, const float* __restrict__ rthr,
               const unsigned short* __restrict__ wvF,
               const unsigned short* __restrict__ w1F,
               const unsigned short* __restrict__ w2F,
               float* __restrict__ gzpart, float* __restrict__ out) {
  extern __shared__ float smem[];
  float*          xs    = smem + SM_XS;
  float*          vs    = smem + SM_VS;
  unsigned short* xbf   = (unsigned short*)(smem + SM_XBF);
  unsigned short* hbf   = (unsigned short*)(smem + SM_HBF);
  float*          psum  = smem + SM_PSUM;
  float*          gatev = smem + SM_GATE;
  float*          gzv   = smem + SM_GZ;

  const int tid   = threadIdx.x;
  const int lane  = tid & 31;
  const int wave  = tid >> 5;
  const int rt    = wave >> 2;         // 0..1  (row tile)
  const int nt0   = (wave & 3) * 8;    // 0,8,16,24 (first N tile)
  const int wg    = blockIdx.x;
  const int rowg0 = wg * ROWS;

  const float decay  = fminf(fmaxf(rdecay[0], 0.5f), 0.99f);
  const float dt     = fminf(fmaxf(rdt[0], 0.01f), 0.5f);
  const float spring = fabsf(rspring[0]) + 0.01f;
  const float thr    = fabsf(rthr[0]) + 0.0001f;

  // ---- load x0 tile into LDS (f32 + bf16 shadow); emit positions[0] -------
  {
    const float4* src  = (const float4*)(x0g + (long long)rowg0 * D);
    float4*       pdst = (float4*)(out + POS_OFF + (long long)rowg0 * D);
    float4*       xsd  = (float4*)xs;
#pragma unroll
    for (int j = 0; j < 16; ++j) {
      const int e4 = tid + j * NTHREADS;          // float4 index over 32x512
      const float4 v = src[e4];
      xsd[e4]  = v;
      pdst[e4] = v;
      const int m = e4 >> 7;                      // 128 float4 per row
      const int c = (e4 & 127) << 2;
      unsigned int p0 = (unsigned int)f2bf(v.x) | ((unsigned int)f2bf(v.y) << 16);
      unsigned int p1 = (unsigned int)f2bf(v.z) | ((unsigned int)f2bf(v.w) << 16);
      *(unsigned int*)(xbf + m * AS + c)     = p0;
      *(unsigned int*)(xbf + m * AS + c + 2) = p1;
    }
  }
  __syncthreads();

  // ---- v0 = x0 @ Wv^T + bv; emit velocities[0] ----------------------------
  {
    v8f acc[8] = {};
    gemm_rowtile(xbf, wvF, rt, nt0, lane, acc);
#pragma unroll
    for (int i = 0; i < 8; ++i) {
      const int n = (nt0 + i) * 16 + (lane & 15);
      const float bvn = bv[n];
#pragma unroll
      for (int r = 0; r < 8; ++r) {
        const int m = rt * 16 + r + ((lane >> 4) << 3);
        const float vv = acc[i][r] + bvn;
        vs[m * D + n] = vv;
        out[VEL_OFF + (long long)(rowg0 + m) * D + n] = vv;
      }
    }
  }
  __syncthreads();

  // ---- 16-step scan -------------------------------------------------------
  for (int t = 0; t < STEPS; ++t) {
    // prefetch this step's 64 KB noise tile behind the GEMMs
    {
      const char* np =
          (const char*)(noise + ((long long)t * BATCH + rowg0) * D);
      __builtin_prefetch(np + tid * 256, 0, 1);
      __builtin_prefetch(np + tid * 256 + 128, 0, 1);
    }

    // per-row ||v||: deterministic 8-partials-per-row tree
    {
      const int r  = tid >> 3;
      const int c0 = (tid & 7) << 6;
      const float* vrow = vs + r * D + c0;
      float s = 0.f;
#pragma unroll 8
      for (int c = 0; c < 64; ++c) { const float v = vrow[c]; s = fmaf(v, v, s); }
      psum[tid] = s;
    }
    __syncthreads();
    if (tid < ROWS) {
      float tot = 0.f;
#pragma unroll
      for (int j = 0; j < 8; ++j) tot += psum[tid * 8 + j];
      const float gz = sqrtf(tot);
      gzv[tid] = gz;
      float a = bg2[0];
#pragma unroll 4
      for (int j = 0; j < 32; ++j) {
        float h = fmaf(gz, Wg1[j], bg1[j]);
        h = fmaxf(h, 0.f);
        a = fmaf(Wg2[j], h, a);
      }
      gatev[tid] = 1.f / (1.f + expf(-a));
    }
    __syncthreads();
    if (tid == 0) {
      float s = 0.f;
      for (int j = 0; j < ROWS; ++j) s += gzv[j];
      gzpart[wg * 32 + (t + 1)] = s;
      if (t == 0) gzpart[wg * 32] = s;   // gray_zones[0] duplicates mean||v0||
    }

    // GEMM1: hbf = tanh(x @ W1a^T + b1)  (bf16, ready as next A operand)
    {
      v8f acc[8] = {};
      gemm_rowtile(xbf, w1F, rt, nt0, lane, acc);
#pragma unroll
      for (int i = 0; i < 8; ++i) {
        const int n = (nt0 + i) * 16 + (lane & 15);
        const float b1n = b1[n];
#pragma unroll
        for (int r = 0; r < 8; ++r) {
          const int m = rt * 16 + r + ((lane >> 4) << 3);
          hbf[m * AS + n] = f2bf(tanhf(acc[i][r] + b1n));
        }
      }
    }
    __syncthreads();

    // GEMM2 + state update + streamed outputs
    {
      v8f acc[8] = {};
      gemm_rowtile(hbf, w2F, rt, nt0, lane, acc);
#pragma unroll
      for (int i = 0; i < 8; ++i) {
        const int n = (nt0 + i) * 16 + (lane & 15);
        const float b2n = b2[n];
#pragma unroll
        for (int r = 0; r < 8; ++r) {
          const int m  = rt * 16 + r + ((lane >> 4) << 3);
          const int gm = rowg0 + m;
          const float force = spring * (acc[i][r] + b2n);
          const float nz = noise[((long long)t * BATCH + gm) * D + n];
          const float vn = fmaf(decay, vs[m * D + n], force) +
                           nz * gatev[m] * 0.1f;
          const float xn = fmaf(dt, vn, xs[m * D + n]);
          vs[m * D + n]   = vn;
          xs[m * D + n]   = xn;
          xbf[m * AS + n] = f2bf(xn);     // refresh bf16 shadow once
          const long long o = (long long)gm * D + n;
          out[POS_OFF + (long long)(t + 1) * BD + o] = xn;
          out[VEL_OFF + (long long)(t + 1) * BD + o] = vn;
        }
      }
    }
    __syncthreads();
  }

  // ---- converged = ||v16|| < threshold ------------------------------------
  {
    const int r  = tid >> 3;
    const int c0 = (tid & 7) << 6;
    const float* vrow = vs + r * D + c0;
    float s = 0.f;
#pragma unroll 8
    for (int c = 0; c < 64; ++c) { const float v = vrow[c]; s = fmaf(v, v, s); }
    psum[tid] = s;
  }
  __syncthreads();
  if (tid < ROWS) {
    float tot = 0.f;
#pragma unroll
    for (int j = 0; j < 8; ++j) tot += psum[tid * 8 + j];
    out[CONV_OFF + rowg0 + tid] = (sqrtf(tot) < thr) ? 1.0f : 0.0f;
  }

  // ---- xf / vf from LDS ---------------------------------------------------
  {
    const float4* xsv = (const float4*)xs;
    const float4* vsv = (const float4*)vs;
    float4* xfd = (float4*)(out + (long long)rowg0 * D);
    float4* vfd = (float4*)(out + BD + (long long)rowg0 * D);
#pragma unroll
    for (int j = 0; j < 16; ++j) {
      xfd[tid + j * NTHREADS] = xsv[tid + j * NTHREADS];
      vfd[tid + j * NTHREADS] = vsv[tid + j * NTHREADS];
    }
  }
}

// ---------------------------------------------------------------------------
// Deterministic cross-WG reduction for gray_zones
// ---------------------------------------------------------------------------
__global__ void finish_gz(const float* __restrict__ part,
                          float* __restrict__ outgz) {
  const int k = threadIdx.x;
  if (k < 17) {
    float s = 0.f;
    for (int w = 0; w < NWG; ++w) s += part[w * 32 + k];
    outgz[k] = s * (1.0f / (float)BATCH);
  }
}

// ---------------------------------------------------------------------------
extern "C" void kernel_launch(void* const* d_in, const int* in_sizes, int n_in,
                              void* d_out, int out_size, void* d_ws,
                              size_t ws_size, hipStream_t stream) {
  (void)in_sizes; (void)n_in; (void)out_size; (void)ws_size;
  const float* x0   = (const float*)d_in[0];
  const float* nz   = (const float*)d_in[1];
  const float* Wv   = (const float*)d_in[2];
  const float* bv   = (const float*)d_in[3];
  const float* W1   = (const float*)d_in[4];
  const float* b1   = (const float*)d_in[5];
  const float* W2   = (const float*)d_in[6];
  const float* b2   = (const float*)d_in[7];
  const float* Wg1  = (const float*)d_in[8];
  const float* bg1  = (const float*)d_in[9];
  const float* Wg2  = (const float*)d_in[10];
  const float* bg2  = (const float*)d_in[11];
  const float* rdec = (const float*)d_in[12];
  const float* rdt  = (const float*)d_in[13];
  const float* rspr = (const float*)d_in[14];
  const float* rthr = (const float*)d_in[15];

  unsigned short* wF = (unsigned short*)d_ws;
  float* gzpart = (float*)((char*)d_ws + GZPART_OFF);
  float* out = (float*)d_out;

  prep_pack<<<(3 * WFRAG_ELEMS) / NTHREADS, NTHREADS, 0, stream>>>(
      Wv, W1, W2, wF);

  const size_t shbytes = (size_t)SM_FLTS * sizeof(float);   // ~194 KB of 320 KB
  reasoning_main<<<NWG, NTHREADS, shbytes, stream>>>(
      x0, nz, bv, b1, b2, Wg1, bg1, Wg2, bg2, rdec, rdt, rspr, rthr,
      wF, wF + WFRAG_ELEMS, wF + 2 * WFRAG_ELEMS, gzpart, out);

  finish_gz<<<1, 32, 0, stream>>>(gzpart, out + GZ_OFF);
}
